// CustomLSTMLogic_44143673869085
// MI455X (gfx1250) — matrix operand order repaired and checked by reading, not compiled, not run on hardware
//
#include <hip/hip_runtime.h>
#include <math.h>

// ---------------------------------------------------------------------------
// Problem constants (match the JAX reference)
// ---------------------------------------------------------------------------
constexpr int BB = 64;     // batch
constexpr int TT = 512;    // sequence length
constexpr int LL = 2;      // layers
constexpr int DD = 1024;   // input dim
constexpr int HH = 1024;   // hidden dim (== DD, both layers have K = 1024)
constexpr int G4H = 4 * HH;

constexpr int LDW  = HH + 8;                       // padded LDS row stride (elems)
constexpr int SMEM = 2 * 64 * LDW * 2;             // Wih tile + Whh tile (bytes)

// ---------------------------------------------------------------------------
// CDNA5 WMMA types
// ---------------------------------------------------------------------------
typedef __bf16 bf16_t;
typedef __attribute__((ext_vector_type(16))) __bf16 v16bf;
typedef __attribute__((ext_vector_type(8)))  float  v8f;

union FragB16 {
    v16bf v;
    uint4 q[2];
};

__device__ __forceinline__ bf16_t f2bf(float x) {
    union { float f; unsigned int u; } v; v.f = x;
    unsigned int r = v.u + 0x7FFFu + ((v.u >> 16) & 1u); // round-nearest-even
    union { unsigned short u; bf16_t b; } o;
    o.u = (unsigned short)(r >> 16);
    return o.b;
}

// A fragment: 16x32 bf16 tile from global, row-major, ld = row stride (elems).
// ISA layout (16-bit A 16x32): lanes 0-15 -> rows, K {0..7, 16..23};
//                              lanes 16-31 -> rows, K {8..15, 24..31}.
__device__ __forceinline__ v16bf load_fragA(const bf16_t* __restrict__ base, int ld) {
    const int lane = threadIdx.x & 31;
    const int row  = lane & 15;
    const int kh   = (lane >> 4) * 8;
    const bf16_t* p = base + (size_t)row * ld + kh;
    FragB16 f;
    f.q[0] = *reinterpret_cast<const uint4*>(p);
    f.q[1] = *reinterpret_cast<const uint4*>(p + 16);
    return f.v;
}

// B fragment: 32x16 bf16 (K x N) read from the LDS-staged weight tile.
// B[k,n] = W[n,k]; lanes 0-15 K {0..7, 16..23}, lanes 16-31 K {8..15, 24..31}.
__device__ __forceinline__ v16bf load_fragB_lds(const bf16_t* base, int ldw) {
    const int lane = threadIdx.x & 31;
    const int n    = lane & 15;
    const int kh   = (lane >> 4) * 8;
    const bf16_t* p = base + n * ldw + kh;
    FragB16 f;
    f.q[0] = *reinterpret_cast<const uint4*>(p);
    f.q[1] = *reinterpret_cast<const uint4*>(p + 16);
    return f.v;
}

__device__ __forceinline__ v8f wmma_bf16(v16bf a, v16bf b, v8f c) {
    return __builtin_amdgcn_wmma_f32_16x16x32_bf16(
        false, a, false, b, (short)0, c, false, false);
}

__device__ __forceinline__ float sigmoidf_(float x) {
    return 1.0f / (1.0f + expf(-x));
}

// ---------------------------------------------------------------------------
// Prep kernels (run every launch; deterministic)
// ---------------------------------------------------------------------------
__global__ void f32_to_bf16_k(const float* __restrict__ in, bf16_t* __restrict__ out, int n) {
    int i = blockIdx.x * blockDim.x + threadIdx.x;
    if (i < n) out[i] = f2bf(in[i]);
}

__global__ void bias_sum_k(const float* __restrict__ a, const float* __restrict__ b,
                           float* __restrict__ out, int n) {
    int i = blockIdx.x * blockDim.x + threadIdx.x;
    if (i < n) out[i] = a[i] + b[i];
}

// input_ [B,T,D] fp32  ->  Xbf [T,B,D] bf16
__global__ void xpose_bt_k(const float* __restrict__ in, bf16_t* __restrict__ out) {
    long i = (long)blockIdx.x * blockDim.x + threadIdx.x;
    const long total = (long)BB * TT * DD;
    if (i < total) {
        int  d  = (int)(i % DD);
        long bt = i / DD;
        int  t  = (int)(bt % TT);
        int  b  = (int)(bt / TT);
        out[((long)t * BB + b) * DD + d] = f2bf(in[i]);
    }
}

// hx/cx are [B, L, H]; gather layer slice into [B, H]
__global__ void init_state_k(const float* __restrict__ hx, const float* __restrict__ cx,
                             bf16_t* __restrict__ h0, float* __restrict__ c0, int layer) {
    int i = blockIdx.x * blockDim.x + threadIdx.x;
    if (i < BB * HH) {
        int h = i % HH, b = i / HH;
        long src = ((long)b * LL + layer) * HH + h;
        h0[i] = f2bf(hx[src]);
        c0[i] = cx[src];
    }
}

__global__ void zero_ints_k(int* p, int n) {
    int i = blockIdx.x * blockDim.x + threadIdx.x;
    if (i < n) p[i] = 0;
}

// ---------------------------------------------------------------------------
// Persistent per-layer LSTM kernel.
//   grid = 64 blocks (one per 16 hidden columns, one per WGP given 256KB LDS),
//   block = 128 threads (4 waves, one per 16 batch rows).
// Stages this block's 64 Wih rows + 64 Whh rows in LDS once, keeps cell state
// in registers, loops over all T timesteps.  The grid-wide barrier is a
// split arrive/wait: the wait for h_{t-1} is overlapped with the (independent)
// x-projection of step t, and the arrive after writing h_t is non-blocking.
// ---------------------------------------------------------------------------
__global__ __launch_bounds__(128)
void lstm_layer_k(const bf16_t* __restrict__ xseq,  // [T,B,1024] bf16
                  const bf16_t* __restrict__ h0,    // [B,H]
                  const float*  __restrict__ c0,    // [B,H] initial cell state
                  const bf16_t* __restrict__ Wih,   // [4H, 1024]
                  const bf16_t* __restrict__ Whh,   // [4H, 1024]
                  const float*  __restrict__ bias,  // [4H] (b_ih + b_hh)
                  float*        __restrict__ out,   // [B,T,H] fp32 or nullptr
                  bf16_t*       __restrict__ hbuf,  // [T,B,H] (full) or [2,B,H] (ping-pong)
                  int           pingpong,           // 0 = full sequence, 1 = ping-pong
                  float*        __restrict__ hfin,  // [B,H] final h or nullptr
                  float*        __restrict__ cfin,  // [B,H] final c or nullptr
                  int*          sync) {             // [4]: counters[2], phase
    const int wave = threadIdx.x >> 5;
    const int lane = threadIdx.x & 31;
    const int m0   = wave * 16;            // batch-row base for this wave
    const int n0   = blockIdx.x * 16;      // hidden-column base for this block

    extern __shared__ bf16_t smem[];
    bf16_t* sWih = smem;                   // [64][LDW]
    bf16_t* sWhh = smem + 64 * LDW;        // [64][LDW]

    // ---- stage the weight tiles (rows g*H + n0 + j, j in 0..15, g in 0..3) ----
    for (int idx = threadIdx.x; idx < 64 * (HH / 8); idx += blockDim.x) {
        const int r = idx >> 7;            // LDS row 0..63
        const int c = idx & 127;           // 8-elem chunk in row
        const size_t wrow = (size_t)((r >> 4) * HH + n0 + (r & 15));
        *reinterpret_cast<uint4*>(sWih + r * LDW + c * 8) =
            *reinterpret_cast<const uint4*>(Wih + wrow * HH + c * 8);
        *reinterpret_cast<uint4*>(sWhh + r * LDW + c * 8) =
            *reinterpret_cast<const uint4*>(Whh + wrow * HH + c * 8);
    }
    __syncthreads();

    // ---- per-lane cell-state registers + biases (C-layout: lane = N, VGPR = M)
    const int col  = lane & 15;
    const int hcol = n0 + col;
    const int mAdd = (lane >> 4) * 8;

    const float bi = bias[0 * HH + hcol];
    const float bf = bias[1 * HH + hcol];
    const float bg = bias[2 * HH + hcol];
    const float bo = bias[3 * HH + hcol];

    float creg[8];
#pragma unroll
    for (int r = 0; r < 8; ++r)
        creg[r] = c0[(size_t)(m0 + mAdd + r) * HH + hcol];

    int* counters = sync;
    int* phase    = sync + 2;
    const int nblk = (int)gridDim.x;

    for (int t = 0; t < TT; ++t) {
        v8f acc[4] = {};                   // i, f, g, o tiles

        // ---- input projection x_t @ Wih^T (independent of h_{t-1}) ----
        const bf16_t* xt = xseq + (size_t)t * BB * HH;
        for (int k = 0; k < HH; k += 32) {
            v16bf a = load_fragA(xt + (size_t)m0 * HH + k, HH);
#pragma unroll
            for (int g = 0; g < 4; ++g) {
                v16bf b = load_fragB_lds(sWih + (g * 16) * LDW + k, LDW);
                acc[g] = wmma_bf16(a, b, acc[g]);
            }
        }

        // ---- wait for h_{t-1} (latency hidden behind x-projection above) ----
        if (t > 0) {
            if (threadIdx.x == 0) {
                while (__atomic_load_n(phase, __ATOMIC_RELAXED) < t)
                    __builtin_amdgcn_s_sleep(1);
            }
            __syncthreads();
            __threadfence();
        }

        const bf16_t* hp = (t == 0)
            ? h0
            : (pingpong ? hbuf + (size_t)((t - 1) & 1) * BB * HH
                        : hbuf + (size_t)(t - 1) * BB * HH);

        // ---- recurrent projection h_{t-1} @ Whh^T ----
        for (int k = 0; k < HH; k += 32) {
            v16bf a = load_fragA(hp + (size_t)m0 * HH + k, HH);
#pragma unroll
            for (int g = 0; g < 4; ++g) {
                v16bf b = load_fragB_lds(sWhh + (g * 16) * LDW + k, LDW);
                acc[g] = wmma_bf16(a, b, acc[g]);
            }
        }

        // ---- fused cell math ----
        bf16_t* hn = pingpong ? hbuf + (size_t)(t & 1) * BB * HH
                              : hbuf + (size_t)t * BB * HH;
        union { v8f v; float f[8]; } ai, af, ag, ao;
        ai.v = acc[0]; af.v = acc[1]; ag.v = acc[2]; ao.v = acc[3];

        const bool last = (t == TT - 1);
#pragma unroll
        for (int r = 0; r < 8; ++r) {
            const int m = m0 + mAdd + r;
            const float iv = sigmoidf_(ai.f[r] + bi);
            const float fv = sigmoidf_(af.f[r] + bf);
            const float gv = tanhf(ag.f[r] + bg);
            const float ov = sigmoidf_(ao.f[r] + bo);

            const float cnew = fv * creg[r] + iv * gv;
            const float hnew = ov * tanhf(cnew);
            creg[r] = cnew;

            const size_t sidx = (size_t)m * HH + hcol;
            hn[sidx] = f2bf(hnew);
            if (out)  out[(size_t)m * TT * HH + (size_t)t * HH + hcol] = hnew;
            if (last && hfin) { hfin[sidx] = hnew; cfin[sidx] = cnew; }
        }

        // ---- non-blocking arrive: publish h_t, bump phase if last block ----
        if (t + 1 < TT) {
            __threadfence();
            __syncthreads();
            if (threadIdx.x == 0) {
                if (atomicAdd(&counters[t & 1], 1) == nblk - 1) {
                    counters[t & 1] = 0;          // safe: reused only at t+2
                    __threadfence();
                    __atomic_store_n(phase, t + 1, __ATOMIC_RELAXED);
                }
            }
            // no spin here: next iteration's x-projection overlaps the barrier
        }
    }
}

// ---------------------------------------------------------------------------
// Host-side orchestration
// ---------------------------------------------------------------------------
extern "C" void kernel_launch(void* const* d_in, const int* in_sizes, int n_in,
                              void* d_out, int out_size, void* d_ws, size_t ws_size,
                              hipStream_t stream) {
    const float* x_in = (const float*)d_in[0];   // [B,T,D]
    const float* hx   = (const float*)d_in[1];   // [B,L,H]
    const float* cx   = (const float*)d_in[2];   // [B,L,H]
    const float* Wih  = (const float*)d_in[3];   // [L,4H,D]
    const float* Whh  = (const float*)d_in[4];   // [L,4H,H]
    const float* bih  = (const float*)d_in[5];   // [L,4H]
    const float* bhh  = (const float*)d_in[6];   // [L,4H]
    float* out = (float*)d_out;                  // [B,T,H] ++ [B,H] ++ [B,H]

    char* ws = (char*)d_ws;
    size_t off = 0;
    auto carve = [&](size_t bytes) -> void* {
        void* p = ws + off;
        off = (off + bytes + 255) & ~(size_t)255;
        return p;
    };
    bf16_t* WihB = (bf16_t*)carve((size_t)LL * G4H * DD * sizeof(bf16_t));
    bf16_t* WhhB = (bf16_t*)carve((size_t)LL * G4H * HH * sizeof(bf16_t));
    float*  bias = (float*) carve((size_t)LL * G4H * sizeof(float));
    bf16_t* Xbf  = (bf16_t*)carve((size_t)TT * BB * DD * sizeof(bf16_t));
    bf16_t* Hseq = (bf16_t*)carve((size_t)TT * BB * HH * sizeof(bf16_t));
    bf16_t* h0   = (bf16_t*)carve((size_t)LL * BB * HH * sizeof(bf16_t));
    float*  c0   = (float*) carve((size_t)LL * BB * HH * sizeof(float));
    bf16_t* hpp  = (bf16_t*)carve((size_t)2 * BB * HH * sizeof(bf16_t));
    int*    sync = (int*)   carve(8 * sizeof(int));

    // ---- prep ----
    {
        int n = LL * G4H * DD;
        f32_to_bf16_k<<<(n + 255) / 256, 256, 0, stream>>>(Wih, WihB, n);
        n = LL * G4H * HH;
        f32_to_bf16_k<<<(n + 255) / 256, 256, 0, stream>>>(Whh, WhhB, n);
        n = LL * G4H;
        bias_sum_k<<<(n + 255) / 256, 256, 0, stream>>>(bih, bhh, bias, n);
        long nt = (long)BB * TT * DD;
        xpose_bt_k<<<(int)((nt + 255) / 256), 256, 0, stream>>>(x_in, Xbf);
        n = BB * HH;
        init_state_k<<<(n + 255) / 256, 256, 0, stream>>>(hx, cx, h0, c0, 0);
        init_state_k<<<(n + 255) / 256, 256, 0, stream>>>(hx, cx, h0 + (size_t)BB * HH,
                                                          c0 + (size_t)BB * HH, 1);
        zero_ints_k<<<1, 8, 0, stream>>>(sync, 8);
    }

    (void)hipFuncSetAttribute((const void*)lstm_layer_k,
                              hipFuncAttributeMaxDynamicSharedMemorySize, SMEM);

    const dim3 grid(HH / 16), blk(128);

    // ---- layer 0: persistent over all T, writes bf16 hidden sequence ----
    lstm_layer_k<<<grid, blk, SMEM, stream>>>(
        Xbf, h0, c0, WihB, WhhB, bias,
        /*out=*/nullptr, Hseq, /*pingpong=*/0,
        /*hfin=*/nullptr, /*cfin=*/nullptr, sync);

    // ---- layer 1: persistent, reads layer-0 sequence, writes fp32 output ----
    float* hfin = out + (size_t)BB * TT * HH;
    float* cfin = hfin + (size_t)BB * HH;
    lstm_layer_k<<<grid, blk, SMEM, stream>>>(
        Hseq,
        h0 + (size_t)BB * HH, c0 + (size_t)BB * HH,
        WihB + (size_t)G4H * DD, WhhB + (size_t)G4H * HH, bias + G4H,
        out, hpp, /*pingpong=*/1,
        hfin, cfin, sync + 4);
}
